// DynoFormerLayer_35150012350775
// MI455X (gfx1250) — compile-verified, matrix operand order checked by reading, not run on hardware
//
#include <hip/hip_runtime.h>
#include <hip/hip_bf16.h>
#include <math.h>

// ---------------------------------------------------------------------------
// DynoFormer layer for MI455X (gfx1250, wave32, WMMA + TDM).
//
// Shapes: B=2, SEQ=512, NVAR=64, DM=128, K=2.
//
// Pipeline (all on `stream`, dependencies via stream order):
//   1) k_embed_gemm : xe[b,t,n] = sum_s x[b,s,n]*W_emb[s,t] + b_emb[t]
//                     -> WMMA f32_16x16x32_f16, one wave per 16x16 tile.
//   2) k_pw         : two DFT bins (f1,f2 read from DEVICE memory) -> softmax
//   3) k_derived    : rank-1 collapse of start_embed+emb_linear+QKV convs:
//                     q[bn,e,t] = xe[b,t,n]*qa[e] + qb[e] (t<DM), qp[e] (pad)
//   4) k_attn       : per-(bn,e) period attention (runtime period), VALU+LDS
//                     softmax; accumulates pw-weighted branch outputs -> oc
//   5) k_out_gemm   : out[bn] = oc[bn] @ Wo^T + bo.  One workgroup per bn:
//                     TDM (tensor_load_to_lds, 6-arg clang-23 form) stages both
//                     128x128 f32 panels into LDS (128 KB of the 320 KB WGP
//                     pool), then 8 waves x 8 tiles x 4 unrolled v_wmma.
// ---------------------------------------------------------------------------

#define BB    2
#define SEQ   512
#define NVAR  64
#define DM    128
#define KTOP  2
#define SCALE 0.08838834764831845f   // 128^-0.5
#define TWO_PI 6.283185307179586f

typedef __attribute__((ext_vector_type(16))) _Float16 v16h;
typedef __attribute__((ext_vector_type(8)))  float    v8f;
typedef __attribute__((ext_vector_type(4)))  unsigned int tdm_v4u;
typedef __attribute__((ext_vector_type(8)))  int          tdm_v8i;
typedef __attribute__((ext_vector_type(4)))  int          tdm_v4i;

// Per-lane K index pattern for 16-bit WMMA A/B operands (ISA 7.12.2):
// lane<16 holds K = {0..7, 16..23}; lane>=16 holds K = {8..15, 24..31}.
__device__ __forceinline__ int wmma_kk(int idx, int hi) {
    return (idx & 7) + 8 * hi + 16 * (idx >> 3);
}

// ---------------------------------------------------------------------------
// 1) xe = W_emb^T @ x  (+ b_emb), per batch.  M=DM(t), N=NVAR(n), K=SEQ(s).
//    grid = (DM/16)*(NVAR/16)*B waves of 32 threads.
// ---------------------------------------------------------------------------
__global__ __launch_bounds__(32)
void k_embed_gemm(const float* __restrict__ x,       // [B,SEQ,NVAR]
                  const float* __restrict__ W_emb,   // [SEQ,DM]
                  const float* __restrict__ b_emb,   // [DM]
                  float* __restrict__ xe)            // [B,DM,NVAR]
{
    int blk = blockIdx.x;
    int tn  = blk % (NVAR / 16); blk /= (NVAR / 16);
    int tt  = blk % (DM / 16);
    int b   = blk / (DM / 16);
    int lane = threadIdx.x;
    int r = lane & 15, hi = lane >> 4;
    int t0 = tt * 16, n0 = tn * 16;
    const float* xb = x + (size_t)b * SEQ * NVAR;

    v8f c = {};
    for (int s0 = 0; s0 < SEQ; s0 += 32) {
        v16h a, bm;
#pragma unroll
        for (int idx = 0; idx < 16; ++idx) {
            int s = s0 + wmma_kk(idx, hi);
            a[idx]  = (_Float16)W_emb[s * DM + (t0 + r)];   // A[t,s] = W_emb[s,t]
            bm[idx] = (_Float16)xb[s * NVAR + (n0 + r)];    // B[s,n] = x[b,s,n]
        }
        c = __builtin_amdgcn_wmma_f32_16x16x32_f16(false, a, false, bm,
                                                   (short)0, c, false, false);
    }
#pragma unroll
    for (int v = 0; v < 8; ++v) {
        int t = t0 + v + 8 * hi;
        int n = n0 + r;
        xe[((size_t)b * DM + t) * NVAR + n] = c[v] + b_emb[t];
    }
}

// ---------------------------------------------------------------------------
// 2) pw[b,k] = softmax over k of mean_n |DFT_{f_k}(xe[b,:,n])|.
//    f1,f2 live in device memory (host must not read them under capture).
// ---------------------------------------------------------------------------
__global__ __launch_bounds__(64)
void k_pw(const float* __restrict__ xe, const int* __restrict__ f1p,
          const int* __restrict__ f2p, float* __restrict__ pw)
{
    __shared__ float red[64];
    __shared__ float amps[KTOP];
    int b = blockIdx.x, n = threadIdx.x;
    int fs[KTOP] = { *f1p, *f2p };
#pragma unroll
    for (int k = 0; k < KTOP; ++k) {
        float w = -TWO_PI * (float)fs[k] / (float)DM;
        float re = 0.f, im = 0.f;
        for (int t = 0; t < DM; ++t) {
            float xv = xe[((size_t)b * DM + t) * NVAR + n];
            float ang = w * (float)t;
            re += xv * __cosf(ang);
            im += xv * __sinf(ang);
        }
        red[n] = sqrtf(re * re + im * im);
        __syncthreads();
        for (int s = 32; s > 0; s >>= 1) {
            if (n < s) red[n] += red[n + s];
            __syncthreads();
        }
        if (n == 0) amps[k] = red[0] * (1.0f / NVAR);
        __syncthreads();
    }
    if (n == 0) {
        float mx = fmaxf(amps[0], amps[1]);
        float e0 = __expf(amps[0] - mx), e1 = __expf(amps[1] - mx);
        float inv = 1.0f / (e0 + e1);
        pw[b * KTOP + 0] = e0 * inv;
        pw[b * KTOP + 1] = e1 * inv;
    }
}

// ---------------------------------------------------------------------------
// 3) Rank-1 collapse:  c1 = Wel@W_start, c2 = Wel@b_start + bel.
//    qa = Wq@c1, qb = Wq@c2 + bq, qp = Wq@bel + bq  (pad rows), same for k,v.
//    der layout: [qa qb qp ka kb kp va vb vp], each DM floats.
// ---------------------------------------------------------------------------
__global__ __launch_bounds__(128)
void k_derived(const float* __restrict__ W_start, const float* __restrict__ b_start,
               const float* __restrict__ Wel, const float* __restrict__ bel,
               const float* __restrict__ Wq, const float* __restrict__ bq,
               const float* __restrict__ Wk, const float* __restrict__ bk,
               const float* __restrict__ Wv, const float* __restrict__ bv,
               float* __restrict__ der)
{
    __shared__ float c1[DM], c2[DM];
    int e = threadIdx.x;
    float s1 = 0.f, s2 = 0.f;
    for (int d = 0; d < DM; ++d) {
        float w = Wel[e * DM + d];
        s1 += w * W_start[d];
        s2 += w * b_start[d];
    }
    c1[e] = s1;
    c2[e] = s2 + bel[e];
    __syncthreads();
    float qa = 0, qb2 = 0, qp = 0, ka = 0, kb2 = 0, kp = 0, va = 0, vb2 = 0, vp = 0;
    for (int d = 0; d < DM; ++d) {
        float wq = Wq[e * DM + d], wk = Wk[e * DM + d], wv = Wv[e * DM + d];
        float be = bel[d], a = c1[d], bb2 = c2[d];
        qa += wq * a;  qb2 += wq * bb2;  qp += wq * be;
        ka += wk * a;  kb2 += wk * bb2;  kp += wk * be;
        va += wv * a;  vb2 += wv * bb2;  vp += wv * be;
    }
    der[0 * DM + e] = qa;  der[1 * DM + e] = qb2 + bq[e];  der[2 * DM + e] = qp + bq[e];
    der[3 * DM + e] = ka;  der[4 * DM + e] = kb2 + bk[e];  der[5 * DM + e] = kp + bk[e];
    der[6 * DM + e] = va;  der[7 * DM + e] = vb2 + bv[e];  der[8 * DM + e] = vp + bv[e];
}

// ---------------------------------------------------------------------------
// 4) Period attention per (bn,e); both branches accumulated with pw weights.
//    oc[bn,e,h] = sum_k pw[b,k] * o_k[bn,e,h]   (h = i*pn + p, h < DM)
// ---------------------------------------------------------------------------
__global__ __launch_bounds__(128)
void k_attn(const float* __restrict__ xe, const float* __restrict__ der,
            const float* __restrict__ pw, const int* __restrict__ f1p,
            const int* __restrict__ f2p, float* __restrict__ oc)
{
    int id = blockIdx.x;
    int e  = id % DM;
    int bn = id / DM;
    int b  = bn / NVAR;
    int n  = bn % NVAR;
    int tid = threadIdx.x;

    __shared__ float sQ[256], sK[256], sV[256];
    __shared__ float sAttn[128], sOc[128], sRed[128];

    float qa = der[0 * DM + e], qb = der[1 * DM + e], qp = der[2 * DM + e];
    float ka = der[3 * DM + e], kb = der[4 * DM + e], kp = der[5 * DM + e];
    float va = der[6 * DM + e], vb = der[7 * DM + e], vp = der[8 * DM + e];

    sOc[tid] = 0.f;
    int fs[KTOP] = { *f1p, *f2p };

    for (int kk = 0; kk < KTOP; ++kk) {
        int f = fs[kk];
        if (f < 1) f = 1;                      // safety (fl[0]=0 in reference)
        int P   = DM / f;                      // period
        int rem = DM % P;
        int pad = rem ? (P - rem) : 0;
        int L   = DM + pad;                    // pn * P, L < 256
        int pn  = L / P;
        float w = pw[b * KTOP + kk];

        __syncthreads();                       // sQ/sK/sV reuse across branches
        for (int t = tid; t < L; t += 128) {
            if (t < DM) {
                float xv = xe[((size_t)b * DM + t) * NVAR + n];
                sQ[t] = xv * qa + qb;
                sK[t] = xv * ka + kb;
                sV[t] = xv * va + vb;
            } else {                           // zero-padded x4 -> bel-only rows
                sQ[t] = qp; sK[t] = kp; sV[t] = vp;
            }
        }
        __syncthreads();

        int imax = (DM + pn - 1) / pn;         // only rows that survive [:DM]
        if (imax > P) imax = P;

        for (int i = 0; i < imax; ++i) {
            // logits over j (= tid) for row i, contraction over p (pn terms)
            float lg = -3.0e38f;
            if (tid < P) {
                float a = 0.f;
                for (int p = 0; p < pn; ++p) a += sQ[p * P + i] * sK[p * P + tid];
                lg = a * SCALE;
            }
            sRed[tid] = lg; __syncthreads();
            for (int s = 64; s > 0; s >>= 1) {
                if (tid < s) sRed[tid] = fmaxf(sRed[tid], sRed[tid + s]);
                __syncthreads();
            }
            float mx = sRed[0]; __syncthreads();
            float ex = (tid < P) ? __expf(lg - mx) : 0.f;
            sRed[tid] = ex; __syncthreads();
            for (int s = 64; s > 0; s >>= 1) {
                if (tid < s) sRed[tid] += sRed[tid + s];
                __syncthreads();
            }
            float inv = 1.0f / sRed[0]; __syncthreads();
            sAttn[tid] = ex * inv; __syncthreads();

            // o[i,p] = sum_j attn[i,j] * v[p*P + j];  h = i*pn + p
            if (tid < pn) {
                int h = i * pn + tid;
                if (h < DM) {
                    float acc = 0.f;
                    for (int j = 0; j < P; ++j) acc += sAttn[j] * sV[tid * P + j];
                    sOc[h] += w * acc;
                }
            }
            __syncthreads();
        }
    }
    __syncthreads();
    oc[((size_t)bn * DM + e) * DM + tid] = sOc[tid];
}

// ---------------------------------------------------------------------------
// 5) out[bn] = oc[bn] @ Wo^T + bo.  One workgroup (8 waves) per bn.
//    TDM stages A = oc[bn] (128x128 f32, 64 KB) and B = Wo (64 KB) into LDS,
//    wave 0 issues s_wait_tensorcnt 0, barrier, then wave w computes e-row w:
//    8 m-tiles x 4 unrolled v_wmma_f32_16x16x32_f16 from LDS.
// ---------------------------------------------------------------------------
__global__ __launch_bounds__(256)
void k_out_gemm(const float* __restrict__ oc, const float* __restrict__ Wo,
                const float* __restrict__ bo, float* __restrict__ out)
{
    __shared__ float sA[DM * DM];   // 64 KB : oc[bn]  (A[e,h])
    __shared__ float sB[DM * DM];   // 64 KB : Wo      (B[h,m] = Wo[m,h])
    int bn   = blockIdx.x;
    int tid  = threadIdx.x;
    int wave = tid >> 5;
    int lane = tid & 31;

#if defined(__gfx1250__) && __has_builtin(__builtin_amdgcn_tensor_load_to_lds)
    if (wave == 0) {
        // Tensor DMA descriptors (CDNA5 ISA).
        // 2D tensor: data_size=4B (code 2), tensor 128x128, tile 128x128,
        // dim0 stride 128 elements; groups 2/3 unused (zero) for 2D.
        unsigned long long ga = (unsigned long long)(uintptr_t)(oc + (size_t)bn * DM * DM);
        unsigned long long gb = (unsigned long long)(uintptr_t)Wo;
        unsigned int ldsA = (unsigned int)(uintptr_t)&sA[0];   // low 32 bits = LDS byte offset
        unsigned int ldsB = (unsigned int)(uintptr_t)&sB[0];
        // group0: [1:0]=count=1 | [63:32]=lds_addr | [120:64]=global_addr | [127:126]=type 2
        tdm_v4u g0a = { 1u, ldsA, (unsigned int)ga,
                        (unsigned int)((ga >> 32) & 0x01FFFFFFu) | (2u << 30) };
        tdm_v4u g0b = { 1u, ldsB, (unsigned int)gb,
                        (unsigned int)((gb >> 32) & 0x01FFFFFFu) | (2u << 30) };
        // group1 words:
        //  w0: mask=0, data_size=2 (4B)          -> 2<<16
        //  w1: atomic_barrier_addr=0, tensor_dim0[15:0]=128 -> 128<<16
        //  w2: tensor_dim0[31:16]=0, tensor_dim1[15:0]=128  -> 128<<16
        //  w3: tensor_dim1[31:16]=0, tile_dim0=128          -> 128<<16
        //  w4: tile_dim1=128, tile_dim2=0                   -> 128
        //  w5: tensor_dim0_stride[31:0]=128
        //  w6: stride0[47:32]=0, stride1[15:0]=0 ; w7: stride1[47:16]=0
        tdm_v8i g1 = { (int)(2u << 16), (int)(128u << 16), (int)(128u << 16),
                       (int)(128u << 16), 128, 128, 0, 0 };
        tdm_v4i gz4 = { 0, 0, 0, 0 };
        tdm_v8i gz8 = { 0, 0, 0, 0, 0, 0, 0, 0 };
        // clang-23 / amdgpu-toolchain 6-arg form:
        //   (v4u group0, v8i group1, v4i, v4i, v8i, i32 cpol)
        __builtin_amdgcn_tensor_load_to_lds(g0a, g1, gz4, gz4, gz8, 0);
        __builtin_amdgcn_tensor_load_to_lds(g0b, g1, gz4, gz4, gz8, 0);
        __builtin_amdgcn_s_wait_tensorcnt(0);
    }
    __syncthreads();
#else
    for (int i = tid; i < DM * DM; i += 256) {
        sA[i] = oc[(size_t)bn * DM * DM + i];
        sB[i] = Wo[i];
    }
    __syncthreads();
#endif

    int r = lane & 15, hi = lane >> 4;
    int e0 = wave * 16;                        // wave w owns e-tile row w
    for (int tm = 0; tm < 8; ++tm) {
        int m0 = tm * 16;
        v8f c = {};
#pragma unroll
        for (int h0 = 0; h0 < DM; h0 += 32) {  // 4 x v_wmma, fully unrolled
            v16h a, bm;
#pragma unroll
            for (int idx = 0; idx < 16; ++idx) {
                int k = h0 + wmma_kk(idx, hi);
                a[idx]  = (_Float16)sA[(e0 + r) * DM + k];   // A[e,h] = oc[bn,e,h]
                bm[idx] = (_Float16)sB[(m0 + r) * DM + k];   // B[h,m] = Wo[m,h]
            }
            c = __builtin_amdgcn_wmma_f32_16x16x32_f16(false, a, false, bm,
                                                       (short)0, c, false, false);
        }
#pragma unroll
        for (int v = 0; v < 8; ++v) {
            int e = e0 + v + 8 * hi;
            int m = m0 + r;
            out[((size_t)bn * DM + e) * DM + m] = c[v] + bo[m];
        }
    }
}

// ---------------------------------------------------------------------------
extern "C" void kernel_launch(void* const* d_in, const int* in_sizes, int n_in,
                              void* d_out, int out_size, void* d_ws, size_t ws_size,
                              hipStream_t stream)
{
    const float* x       = (const float*)d_in[0];
    const float* W_emb   = (const float*)d_in[1];
    const float* b_emb   = (const float*)d_in[2];
    const float* W_start = (const float*)d_in[3];
    const float* b_start = (const float*)d_in[4];
    const float* Wel     = (const float*)d_in[5];
    const float* bel     = (const float*)d_in[6];
    const float* Wq      = (const float*)d_in[7];
    const float* bq      = (const float*)d_in[8];
    const float* Wk      = (const float*)d_in[9];
    const float* bk      = (const float*)d_in[10];
    const float* Wv      = (const float*)d_in[11];
    const float* bv      = (const float*)d_in[12];
    const float* Wo      = (const float*)d_in[13];
    const float* bo      = (const float*)d_in[14];
    const int*   f1      = (const int*)d_in[15];
    const int*   f2      = (const int*)d_in[16];

    // Workspace layout (float offsets): all intermediates fully overwritten
    // every call (0xAA poison safe).
    float* ws  = (float*)d_ws;
    float* xe  = ws;            // 16384 : [B,DM,NVAR]
    float* pw  = ws + 16384;    //     4 : [B,K]
    float* der = ws + 16416;    //  1152 : 9 x [DM]
    float* oc  = ws + 32768;    // 2097152 : [B*NVAR, DM, DM]

    k_embed_gemm<<<(DM / 16) * (NVAR / 16) * BB, 32, 0, stream>>>(x, W_emb, b_emb, xe);
    k_pw<<<BB, 64, 0, stream>>>(xe, f1, f2, pw);
    k_derived<<<1, 128, 0, stream>>>(W_start, b_start, Wel, bel,
                                     Wq, bq, Wk, bk, Wv, bv, der);
    k_attn<<<BB * NVAR * DM, 128, 0, stream>>>(xe, der, pw, f1, f2, oc);
    k_out_gemm<<<BB * NVAR, 256, 0, stream>>>(oc, Wo, bo, (float*)d_out);
}
